// ModifiedCLIPSurgery_77472620085678
// MI455X (gfx1250) — matrix operand order repaired and checked by reading, not compiled, not run on hardware
//
#include <hip/hip_runtime.h>
#include <hip/hip_bf16.h>

#define DI __device__ __forceinline__

typedef __attribute__((ext_vector_type(16))) __bf16 v16bf;
typedef __attribute__((ext_vector_type(8)))  float  v8f;
typedef __attribute__((ext_vector_type(4)))  unsigned int v4u;
typedef __attribute__((ext_vector_type(8)))  int v8i;
typedef __attribute__((ext_vector_type(4)))  int v4i;

// Problem constants
constexpr int Bb    = 8;
constexpr int Nn    = 1025;
constexpr int Cc    = 1024;
constexpr int Hh    = 16;
constexpr int HD    = 64;
constexpr int Mrows = Bb * Nn;     // 8200
constexpr int NP    = 1056;        // padded seq len: 66*16 = 33*32
constexpr int QKV_N = 3 * Cc;      // 3072
constexpr int ROWS  = 32;          // query rows per attention workgroup

// GEMM blocking
constexpr int BM = 128;            // workgroup rows
constexpr int BK = 32;             // K chunk (one wmma K)
// each of 8 waves computes 32 rows x 64 cols (2x4 accumulators)

#if defined(__has_builtin)
#  if __has_builtin(__builtin_amdgcn_tensor_load_to_lds)
#    define HAVE_TDM 1
#  endif
#endif
#ifndef HAVE_TDM
#  define HAVE_TDM 0
#endif

union Frag16 { v16bf v; uint4 q[2]; };

// B-fragment (32x16): lane = output column, lanes>=16 hold K 16..31.
DI v16bf load_frag_contig(const __bf16* p) {
  Frag16 f;
  const uint4* q = reinterpret_cast<const uint4*>(p);
  f.q[0] = q[0];
  f.q[1] = q[1];
  return f.v;
}

// A-fragment (16x32): lane&15 = row; lanes>=16 hold K 16..31 (khalf = (lane>>4)*8).
DI v16bf load_frag_a(const __bf16* rowk, int khalf) {
  Frag16 f;
  f.q[0] = *reinterpret_cast<const uint4*>(rowk + 2 * khalf);
  f.q[1] = *reinterpret_cast<const uint4*>(rowk + 2 * khalf + 8);
  return f.v;
}

DI v8f wmma_bf16(v16bf a, v16bf b, v8f c) {
  return __builtin_amdgcn_wmma_f32_16x16x32_bf16(
      false, a, false, b, (short)0, c, false, false);
}

// ---------------------------------------------------------------- TDM
// 2D tensor-load descriptor per CDNA5 ISA:
//   group0: count=1 | lds_addr | global_addr | type=2
//   group1: data_size=1(2B), tensor_dim0=tileW, tensor_dim1=tensorH (rows
//           beyond it read zero), tile_dim0=tileW, tile_dim1=tileH,
//           tensor_dim0_stride=strideElems
DI void tdm_load_2d(unsigned ldsOff, const __bf16* g, int tileW, int tileH,
                    int tensorH, int strideElems) {
#if HAVE_TDM
  unsigned long long ga = (unsigned long long)g;
  v4u g0 = { 1u, ldsOff, (unsigned)ga,
             (unsigned)((ga >> 32) & 0x1FFFFFFull) | (2u << 30) };
  v8i g1 = { 0x10000,                                    // data_size=1 -> 2B
             (tileW & 0xFFFF) << 16,                     // tensor_dim0 lo16
             (tensorH & 0xFFFF) << 16,                   // tensor_dim1 lo16
             ((tensorH >> 16) & 0xFFFF) | ((tileW & 0xFFFF) << 16), // dim1 hi | tile_dim0
             tileH & 0xFFFF,                             // tile_dim1 (tile_dim2=0)
             strideElems,                                // tensor_dim0_stride lo32
             0, 0 };
  v4i z = {};
#if __clang_major__ >= 23
  v8i z8 = {};
  __builtin_amdgcn_tensor_load_to_lds(g0, g1, z, z, z8, 0);
#else
  __builtin_amdgcn_tensor_load_to_lds(g0, g1, z, z, 0);
#endif
#endif
}

DI void stage_fallback(__bf16* lds, const __bf16* g, int tileH, int tensorH,
                       int strideElems, int tid) {
  for (int i = tid; i < tileH * BK; i += 256) {
    int row = i >> 5, col = i & 31;
    lds[i] = (row < tensorH) ? g[(long)row * strideElems + col] : (__bf16)0.f;
  }
}

// ---------------------------------------------------------------- converts
__global__ void cvt_f32_bf16(const float* __restrict__ in, __bf16* __restrict__ out, long n) {
  long i = (long)blockIdx.x * blockDim.x + threadIdx.x;
  long stride = (long)gridDim.x * blockDim.x;
  for (; i < n; i += stride) out[i] = (__bf16)in[i];
}

// ---------------------------------------------------------------- GEMM core
// Workgroup computes a BM x 128 block of  out = X @ W^T  (K = Cc).
// A-panel [128][32] and B-panel [128][32] staged in LDS via TDM per K chunk.
struct Accs { v8f a[2][4]; };

DI void gemm_block(const __bf16* __restrict__ X, const __bf16* __restrict__ W,
                   int mtBase, int ntBase, int tensorHA, int tensorHB,
                   __bf16* aT, __bf16* bT, Accs& acc) {
  const int tid  = threadIdx.x;
  const int lane = tid & 31;
  const int wid  = tid >> 5;
  const int wr   = wid >> 1;     // 0..3 : 32-row strip
  const int wc   = wid & 1;      // 0..1 : 64-col strip
  const int khalf   = (lane >> 4) * 8;
  const int khalf16 = (lane & 16);

  for (int s = 0; s < 2; ++s)
    for (int t = 0; t < 4; ++t)
      acc.a[s][t] = (v8f){};

#if HAVE_TDM
  const unsigned ldsA = (unsigned)(unsigned long long)(void*)aT;
  const unsigned ldsB = (unsigned)(unsigned long long)(void*)bT;
#endif

  for (int kc = 0; kc < Cc; kc += BK) {
    __syncthreads();   // previous chunk's LDS reads done before overwrite
#if HAVE_TDM
    if (wid == 0) {
      tdm_load_2d(ldsA, X + (long)mtBase * Cc + kc, BK, BM, tensorHA, Cc);
      tdm_load_2d(ldsB, W + (long)ntBase * Cc + kc, BK, BM, tensorHB, Cc);
      __builtin_amdgcn_s_wait_tensorcnt(0);
    }
#else
    stage_fallback(aT, X + (long)mtBase * Cc + kc, BM, tensorHA, Cc, tid);
    stage_fallback(bT, W + (long)ntBase * Cc + kc, BM, tensorHB, Cc, tid);
#endif
    __syncthreads();

    v16bf af[2], bf[4];
    for (int s = 0; s < 2; ++s)
      af[s] = load_frag_a(aT + (wr * 32 + s * 16 + (lane & 15)) * BK, khalf);
    for (int t = 0; t < 4; ++t)
      bf[t] = load_frag_contig(bT + (wc * 64 + t * 16 + (lane & 15)) * BK + khalf16);
    for (int s = 0; s < 2; ++s)
      for (int t = 0; t < 4; ++t)
        acc.a[s][t] = wmma_bf16(af[s], bf[t], acc.a[s][t]);
  }
}

// ---------------------------------------------------------------- QKV GEMM
__global__ void qkv_gemm_kernel(const __bf16* __restrict__ X, const __bf16* __restrict__ W,
                                __bf16* __restrict__ Qb, __bf16* __restrict__ Kb,
                                __bf16* __restrict__ Vb) {
  __shared__ __align__(16) __bf16 aT[BM * BK];
  __shared__ __align__(16) __bf16 bT[BM * BK];
  const int ntBase = blockIdx.x * 128;
  const int mtBase = blockIdx.y * BM;
  Accs acc;
  gemm_block(X, W, mtBase, ntBase, Mrows - mtBase, QKV_N - ntBase, aT, bT, acc);

  const int lane = threadIdx.x & 31;
  const int wid  = threadIdx.x >> 5;
  const int wr = wid >> 1, wc = wid & 1;
  const int hi8 = (lane >> 4) * 8;
  for (int s = 0; s < 2; ++s) {
    for (int t = 0; t < 4; ++t) {
      const int j = ntBase + wc * 64 + t * 16 + (lane & 15);
      const int which = j >> 10;            // uniform across 16-col tile
      const int c = j & (Cc - 1);
      const int h = c >> 6, d = c & 63;
      __bf16* dst = which == 0 ? Qb : (which == 1 ? Kb : Vb);
      for (int r = 0; r < 8; ++r) {
        int m = mtBase + wr * 32 + s * 16 + r + hi8;
        if (m < Mrows) {
          int bidx = m / Nn, nn = m % Nn;
          dst[((long)(bidx * Hh + h) * NP + nn) * HD + d] = (__bf16)acc.a[s][t][r];
        }
      }
    }
  }
}

// ---------------------------------------------------------------- projection
__global__ void proj_gemm_kernel(const __bf16* __restrict__ Xa, const __bf16* __restrict__ W,
                                 const float* __restrict__ bias, float* __restrict__ out) {
  __shared__ __align__(16) __bf16 aT[BM * BK];
  __shared__ __align__(16) __bf16 bT[BM * BK];
  const int ntBase = blockIdx.x * 128;
  const int mtBase = blockIdx.y * BM;
  Accs acc;
  gemm_block(Xa, W, mtBase, ntBase, Mrows - mtBase, Cc - ntBase, aT, bT, acc);

  const int lane = threadIdx.x & 31;
  const int wid  = threadIdx.x >> 5;
  const int wr = wid >> 1, wc = wid & 1;
  const int hi8 = (lane >> 4) * 8;
  for (int s = 0; s < 2; ++s) {
    for (int t = 0; t < 4; ++t) {
      const int j = ntBase + wc * 64 + t * 16 + (lane & 15);
      const float bj = bias[j];
      for (int r = 0; r < 8; ++r) {
        int m = mtBase + wr * 32 + s * 16 + r + hi8;
        if (m < Mrows) out[(long)m * Cc + j] = acc.a[s][t][r] + bj;
      }
    }
  }
}

// ---------------------------------------------------------------- attention
// One workgroup = (b, h, 32-row strip). Path 0: QK^T softmax @ V -> Xori.
// Path 1: VV^T softmax @ V -> Xnew. Scores kept in LDS (32 x 1056 fp32).
__global__ void attn_kernel(const __bf16* __restrict__ Qb, const __bf16* __restrict__ Kb,
                            const __bf16* __restrict__ Vb,
                            __bf16* __restrict__ Xnew, __bf16* __restrict__ Xori) {
  extern __shared__ float smem[];
  float* sc     = smem;                 // ROWS*NP
  float* red    = sc + ROWS * NP;       // ROWS*8
  float* rowmax = red + ROWS * 8;       // ROWS
  float* rowsum = rowmax + ROWS;        // ROWS

  const int tid  = threadIdx.x;
  const int lane = tid & 31;
  const int wid  = tid >> 5;
  const int rt = blockIdx.x, h = blockIdx.y, b = blockIdx.z;
  const int rowbase = rt * ROWS;
  const long headoff = (long)(b * Hh + h) * NP * HD;
  const __bf16* Qh = Qb + headoff;
  const __bf16* Kh = Kb + headoff;
  const __bf16* Vh = Vb + headoff;
  const int khalf   = (lane >> 4) * 8;
  const int khalf16 = (lane & 16);
  const float scale = 0.125f;  // 64^-0.5

  for (int p = 0; p < 2; ++p) {
    const __bf16* Ah = (p == 0) ? Qh : Vh;
    const __bf16* Bh = (p == 0) ? Kh : Vh;
    __bf16* dst = (p == 0) ? Xori : Xnew;

    v16bf af[2][2];
    for (int s = 0; s < 2; ++s) {
      const __bf16* arow = Ah + (long)(rowbase + s * 16 + (lane & 15)) * HD;
      af[s][0] = load_frag_a(arow, khalf);
      af[s][1] = load_frag_a(arow + 32, khalf);
    }

    // scores: 2 subtiles x 66 column tiles over 8 waves
    for (int t = wid; t < 2 * (NP / 16); t += 8) {
      const int ct = t >> 1, s = t & 1;
      const __bf16* brow = Bh + (long)(ct * 16 + (lane & 15)) * HD + khalf16;
      __builtin_prefetch(brow + 8 * 16 * HD, 0, 1);   // tile two iterations ahead for this wave
      v8f acc = {};
      acc = wmma_bf16(af[s][0], load_frag_contig(brow), acc);
      acc = wmma_bf16(af[s][1], load_frag_contig(brow + 32), acc);
      const int col = ct * 16 + (lane & 15);
      const int rl  = s * 16 + ((lane >> 4) * 8);
      for (int r = 0; r < 8; ++r)
        sc[(long)(rl + r) * NP + col] = acc[r] * scale;
    }
    __syncthreads();

    // row max over valid columns (8 threads per row)
    {
      const int r = tid >> 3, g = tid & 7;
      float mx = -3.4e38f;
      for (int c2 = g; c2 < Nn; c2 += 8) mx = fmaxf(mx, sc[(long)r * NP + c2]);
      red[r * 8 + g] = mx;
    }
    __syncthreads();
    if ((tid & 7) == 0) {
      const int r = tid >> 3;
      float mx = red[r * 8];
      for (int g = 1; g < 8; ++g) mx = fmaxf(mx, red[r * 8 + g]);
      rowmax[r] = mx;
    }
    __syncthreads();
    // exponentiate (mask padded columns) + row sums
    {
      const int r = tid >> 3, g = tid & 7;
      const float mx = rowmax[r];
      float sm = 0.f;
      for (int c2 = g; c2 < NP; c2 += 8) {
        float pv = 0.f;
        if (c2 < Nn) pv = __expf(sc[(long)r * NP + c2] - mx);
        sc[(long)r * NP + c2] = pv;
        sm += pv;
      }
      red[r * 8 + g] = sm;
    }
    __syncthreads();
    if ((tid & 7) == 0) {
      const int r = tid >> 3;
      float sm = 0.f;
      for (int g = 0; g < 8; ++g) sm += red[r * 8 + g];
      rowsum[r] = sm;
    }
    __syncthreads();

    // P @ V: 8 waves = 2 row-subtiles x 4 d-tiles; K = NP in 33 chunks of 32
    {
      const int s = wid >> 2, dt = wid & 3;
      const int prow = s * 16 + (lane & 15);
      const int dcol = dt * 16 + (lane & 15);
      v8f acc = {};
      for (int kb = 0; kb < NP; kb += 32) {
        __builtin_prefetch(Vh + (long)(kb + 32 + khalf16) * HD + dcol, 0, 1);
        v16bf a;
        for (int e = 0; e < 16; ++e) {
          int kk = kb + 2 * khalf + e;
          a[e] = (__bf16)sc[(long)prow * NP + kk];
        }
        v16bf bv;
        for (int e = 0; e < 16; ++e) {
          int kk = kb + khalf16 + e;
          bv[e] = Vh[(long)kk * HD + dcol];
        }
        acc = wmma_bf16(a, bv, acc);
      }
      const int hi8 = (lane >> 4) * 8;
      for (int r = 0; r < 8; ++r) {
        const int rl = s * 16 + r + hi8;
        const int nglob = rowbase + rl;
        if (nglob < Nn) {
          const float inv = 1.0f / rowsum[rl];
          dst[((long)(b * Nn + nglob)) * Cc + h * HD + dcol] = (__bf16)(acc[r] * inv);
        }
      }
    }
    __syncthreads();
  }
}

// ---------------------------------------------------------------- launch
extern "C" void kernel_launch(void* const* d_in, const int* in_sizes, int n_in,
                              void* d_out, int out_size, void* d_ws, size_t ws_size,
                              hipStream_t stream) {
  (void)in_sizes; (void)n_in; (void)out_size; (void)ws_size;
  const float* x      = (const float*)d_in[0];
  const float* w_qkv  = (const float*)d_in[1];
  const float* w_proj = (const float*)d_in[2];
  const float* b_proj = (const float*)d_in[3];
  float* out = (float*)d_out;

  char* ws = (char*)d_ws;
  size_t off = 0;
  auto alloc = [&](size_t bytes) -> void* {
    void* p = ws + off;
    off += (bytes + 255) & ~(size_t)255;
    return p;
  };
  __bf16* xb  = (__bf16*)alloc((size_t)Mrows * Cc * 2);
  __bf16* wqb = (__bf16*)alloc((size_t)QKV_N * Cc * 2);
  __bf16* wpb = (__bf16*)alloc((size_t)Cc * Cc * 2);
  const size_t qkvElems = (size_t)Bb * Hh * NP * HD;
  __bf16* Qb = (__bf16*)alloc(qkvElems * 2);
  __bf16* Kb = (__bf16*)alloc(qkvElems * 2);
  __bf16* Vb = (__bf16*)alloc(qkvElems * 2);
  __bf16* xnew = (__bf16*)alloc((size_t)Mrows * Cc * 2);
  __bf16* xori = (__bf16*)alloc((size_t)Mrows * Cc * 2);

  // zero Q/K/V (contiguous, 256B-aligned sizes): padded rows give exact-0 scores
  hipMemsetAsync(Qb, 0, qkvElems * 2 * 3, stream);

  cvt_f32_bf16<<<2048, 256, 0, stream>>>(x,      xb,  (long)Mrows * Cc);
  cvt_f32_bf16<<<2048, 256, 0, stream>>>(w_qkv,  wqb, (long)QKV_N * Cc);
  cvt_f32_bf16<<<1024, 256, 0, stream>>>(w_proj, wpb, (long)Cc * Cc);

  const int mBlocks = (Mrows + BM - 1) / BM;     // 65
  qkv_gemm_kernel<<<dim3(QKV_N / 128, mBlocks), 256, 0, stream>>>(xb, wqb, Qb, Kb, Vb);

  const size_t smemBytes = ((size_t)ROWS * NP + ROWS * 8 + 2 * ROWS) * sizeof(float);
  hipFuncSetAttribute((const void*)attn_kernel,
                      hipFuncAttributeMaxDynamicSharedMemorySize, (int)smemBytes);
  attn_kernel<<<dim3(NP / ROWS, Hh, Bb), 256, smemBytes, stream>>>(Qb, Kb, Vb, xnew, xori);

  // reference returns (out_vv, out_ori): VV path first
  proj_gemm_kernel<<<dim3(Cc / 128, mBlocks), 256, 0, stream>>>(xnew, wpb, b_proj, out);
  proj_gemm_kernel<<<dim3(Cc / 128, mBlocks), 256, 0, stream>>>(xori, wpb, b_proj,
                                                                out + (size_t)Mrows * Cc);
}
